// SimCLR_Loss_29334626631754
// MI455X (gfx1250) — matrix operand order repaired and checked by reading, not compiled, not run on hardware
//
#include <hip/hip_runtime.h>
#include <hip/hip_bf16.h>

// SimCLR loss: fused WMMA Gram-matrix + streaming sum-exp.
// - zn pre-scaled by sqrt(2*log2e) so WMMA accumulator holds d = (sim/T)*log2e
//   => exp(sim/T) = exp2(d) (single v_exp_f32), pos = d_pos * ln2 at the end.
// - B tiles staged block-wide through LDS via global_load_async_to_lds_b128
//   (ASYNCcnt), double-buffered, shared by 8 waves => 8x less L2 traffic.

typedef __attribute__((ext_vector_type(16))) _Float16 v16h;
typedef __attribute__((ext_vector_type(8)))  _Float16 v8h;
typedef __attribute__((ext_vector_type(8)))  float    v8f;

#define SIMCLR_N 8192
#define SIMCLR_B 4096
#define SIMCLR_D 256
#define N_CHUNKS 4
#define TILES_PER_CHUNK ((SIMCLR_N / 16) / N_CHUNKS)     // 128
#define TILES_PER_STAGE 4
#define NSTAGES (TILES_PER_CHUNK / TILES_PER_STAGE)      // 32
#define STAGE_HALFS (TILES_PER_STAGE * 16 * SIMCLR_D)    // 16384 halfs = 32 KB
#define STAGE_BYTES (STAGE_HALFS * 2)

// Raw 32-bit LDS byte offset of a __shared__ pointer (for async-DMA VDST).
__device__ __forceinline__ unsigned lds_offset(const void* p) {
  return (unsigned)(uintptr_t)(__attribute__((address_space(3))) const char*)p;
}

// Async-copy one 32 KB stage (4 column tiles, contiguous in zn) into LDS.
// 256 threads x 8 transfers x 16 B = 32 KB. Tracked by ASYNCcnt.
__device__ __forceinline__ void stage_prefetch(const _Float16* zn, int ctStart,
                                               unsigned ldsbase, int tid) {
  const char* g = (const char*)zn + (size_t)ctStart * (16 * SIMCLR_D * 2);
#pragma unroll
  for (int j = 0; j < 8; ++j) {
    const unsigned loff = ldsbase + (unsigned)((j * 256 + tid) * 16);
    const char*    gp   = g + (j * 256 + tid) * 16;
    asm volatile("global_load_async_to_lds_b128 %0, %1, off"
                 :: "v"(loff), "v"(gp) : "memory");
  }
}

// ---------------------------------------------------------------------------
// Kernel 1: row-normalize z = [z_i; z_j], scale by sqrt(2*log2e), emit f16.
// ---------------------------------------------------------------------------
__global__ __launch_bounds__(256) void simclr_normalize(
    const float* __restrict__ zi, const float* __restrict__ zj,
    _Float16* __restrict__ zn) {
  const int lane = threadIdx.x & 31;
  const int row  = blockIdx.x * 8 + (threadIdx.x >> 5);

  const float* src = (row < SIMCLR_B)
                         ? (zi + (size_t)row * SIMCLR_D)
                         : (zj + (size_t)(row - SIMCLR_B) * SIMCLR_D);

  float4 x0 = ((const float4*)src)[lane * 2 + 0];
  float4 x1 = ((const float4*)src)[lane * 2 + 1];

  float ss = x0.x * x0.x + x0.y * x0.y + x0.z * x0.z + x0.w * x0.w +
             x1.x * x1.x + x1.y * x1.y + x1.z * x1.z + x1.w * x1.w;
#pragma unroll
  for (int off = 16; off > 0; off >>= 1)
    ss += __shfl_xor(ss, off, 32);

  const float SCALE = sqrtf(2.0f * 1.44269504088896340736f);  // sqrt(2*log2e)
  const float inv = rsqrtf(ss) * SCALE;

  v8h o;
  o[0] = (_Float16)(x0.x * inv);
  o[1] = (_Float16)(x0.y * inv);
  o[2] = (_Float16)(x0.z * inv);
  o[3] = (_Float16)(x0.w * inv);
  o[4] = (_Float16)(x1.x * inv);
  o[5] = (_Float16)(x1.y * inv);
  o[6] = (_Float16)(x1.z * inv);
  o[7] = (_Float16)(x1.w * inv);
  *(v8h*)(zn + (size_t)row * SIMCLR_D + lane * 8) = o;
}

// ---------------------------------------------------------------------------
// Kernel 2: block = 8 waves, all on the SAME column chunk; wave w owns rows
// [ (rgBase+w)*32, +32 ). B tiles staged via async-DMA into double-buffered
// LDS (one s_wait_asynccnt + one barrier per 4 tiles); fragments read with
// ds_load_b128; 16 v_wmma_f32_16x16x32_f16 per tile; exp2-only fast epilogue.
// ---------------------------------------------------------------------------
__global__ __launch_bounds__(256) void simclr_main(
    const _Float16* __restrict__ zn,
    float* __restrict__ ws_se, float* __restrict__ ws_pos) {
  __shared__ _Float16 ldsbuf[2 * STAGE_HALFS];             // 64 KB

  const int tid      = threadIdx.x;
  const int lane     = tid & 31;
  const int wave     = tid >> 5;
  const int chunk    = blockIdx.x & (N_CHUNKS - 1);        // same for block
  const int rowGroup = (blockIdx.x >> 2) * 8 + wave;       // 0..255
  const int rowbase  = rowGroup * 32;
  const int halfsel  = lane >> 4;
  const int l15      = lane & 15;

  // ---- Resident A fragments for two 16-row tiles (contiguous k = kk*32 + 16*halfsel + s).
  v16h a0[8], a1[8];
  const _Float16* ar0 = zn + (size_t)(rowbase + l15) * SIMCLR_D;
  const _Float16* ar1 = ar0 + 16 * SIMCLR_D;
#pragma unroll
  for (int kk = 0; kk < 8; ++kk) {
    const int k0 = kk * 32 + halfsel * 16;
    v8h lo0 = *(const v8h*)(ar0 + k0);
    v8h hi0 = *(const v8h*)(ar0 + k0 + 8);
    v8h lo1 = *(const v8h*)(ar1 + k0);
    v8h hi1 = *(const v8h*)(ar1 + k0 + 8);
#pragma unroll
    for (int i = 0; i < 8; ++i) {
      a0[kk][i] = lo0[i]; a0[kk][8 + i] = hi0[i];
      a1[kk][i] = lo1[i]; a1[kk][8 + i] = hi1[i];
    }
  }

  float se0[8], se1[8], p0[8], p1[8];
#pragma unroll
  for (int v = 0; v < 8; ++v) {
    se0[v] = 0.0f; se1[v] = 0.0f; p0[v] = 0.0f; p1[v] = 0.0f;
  }

  // Special column tiles (wave-uniform): diagonals + partner tiles.
  const int dt0 = rowGroup * 2;
  const int dt1 = dt0 + 1;
  const int pt0 = dt0 ^ (SIMCLR_B / 16);
  const int pt1 = dt1 ^ (SIMCLR_B / 16);

  const int ctBeg = chunk * TILES_PER_CHUNK;
  const unsigned ldsbase0 = lds_offset(&ldsbuf[0]);

  // Prime the pipeline: stage 0 -> buffer 0.
  stage_prefetch(zn, ctBeg, ldsbase0, tid);

  for (int s = 0; s < NSTAGES; ++s) {
    const int d = s & 1;
    // Own async copies for current buffer complete, then block-wide sync.
    asm volatile("s_wait_asynccnt 0x0" ::: "memory");
    __syncthreads();
    // Kick off next stage into the other buffer (hidden under compute).
    if (s + 1 < NSTAGES)
      stage_prefetch(zn, ctBeg + (s + 1) * TILES_PER_STAGE,
                     ldsbase0 + (unsigned)((d ^ 1) * STAGE_BYTES), tid);

    const _Float16* buf = &ldsbuf[d * STAGE_HALFS];
    for (int t = 0; t < TILES_PER_STAGE; ++t) {
      const int ct = ctBeg + s * TILES_PER_STAGE + t;
      // B fragment from LDS: lane holds column N = ct*16 + l15,
      // halfs j -> K = kk*32 + 16*halfsel + j (contiguous 32 B).
      const _Float16* bcol =
          buf + t * (16 * SIMCLR_D) + l15 * SIMCLR_D + halfsel * 16;

      v8f c0 = {0.f, 0.f, 0.f, 0.f, 0.f, 0.f, 0.f, 0.f};
      v8f c1 = {0.f, 0.f, 0.f, 0.f, 0.f, 0.f, 0.f, 0.f};
#pragma unroll
      for (int kk = 0; kk < 8; ++kk) {
        v16h b = *(const v16h*)(bcol + kk * 32);
        c0 = __builtin_amdgcn_wmma_f32_16x16x32_f16(
            false, a0[kk], false, b, (short)0, c0, false, false);
        c1 = __builtin_amdgcn_wmma_f32_16x16x32_f16(
            false, a1[kk], false, b, (short)0, c1, false, false);
      }

      if (ct == dt0 || ct == dt1 || ct == pt0 || ct == pt1) {
        // Slow path (<= 4 of 128 tiles): diagonal mask + positive grab.
        const int gcol = ct * 16 + l15;
#pragma unroll
        for (int v = 0; v < 8; ++v) {
          const int grow0 = rowbase + v + halfsel * 8;
          const int grow1 = grow0 + 16;
          const float e0 = __builtin_amdgcn_exp2f(c0[v]);
          const float e1 = __builtin_amdgcn_exp2f(c1[v]);
          if (gcol != grow0)               se0[v] += e0;
          if (gcol != grow1)               se1[v] += e1;
          if (gcol == (grow0 ^ SIMCLR_B))  p0[v]  += c0[v];
          if (gcol == (grow1 ^ SIMCLR_B))  p1[v]  += c1[v];
        }
      } else {
        // Fast path: exp2 + add only.
#pragma unroll
        for (int v = 0; v < 8; ++v) {
          se0[v] += __builtin_amdgcn_exp2f(c0[v]);
          se1[v] += __builtin_amdgcn_exp2f(c1[v]);
        }
      }
    }
  }

  // ---- Reduce each row across its 16 lanes (xor 1,2,4,8 stays in-half).
#pragma unroll
  for (int v = 0; v < 8; ++v) {
    float s0 = se0[v], s1 = se1[v], q0 = p0[v], q1 = p1[v];
#pragma unroll
    for (int off = 1; off < 16; off <<= 1) {
      s0 += __shfl_xor(s0, off, 32);
      s1 += __shfl_xor(s1, off, 32);
      q0 += __shfl_xor(q0, off, 32);
      q1 += __shfl_xor(q1, off, 32);
    }
    se0[v] = s0; se1[v] = s1; p0[v] = q0; p1[v] = q1;
  }

  // lane 0 holds rows +0..7, lane 16 rows +8..15 of each 16-row tile.
  if (l15 == 0) {
    const int base = chunk * SIMCLR_N + rowbase + 8 * halfsel;
#pragma unroll
    for (int v = 0; v < 8; ++v) {
      ws_se[base + v]       = se0[v];
      ws_pos[base + v]      = p0[v];
      ws_se[base + 16 + v]  = se1[v];
      ws_pos[base + 16 + v] = p1[v];
    }
  }
}

// ---------------------------------------------------------------------------
// Kernel 3: deterministic combine: per row sum chunk partials, log2, minus
// pos (log2e units), * ln2 / N at the very end.
// ---------------------------------------------------------------------------
__global__ __launch_bounds__(256) void simclr_finalize(
    const float* __restrict__ ws_se, const float* __restrict__ ws_pos,
    float* __restrict__ out) {
  __shared__ float red[8];
  float acc = 0.0f;
  for (int row = threadIdx.x; row < SIMCLR_N; row += 256) {
    float se = 0.0f, pd = 0.0f;
#pragma unroll
    for (int ch = 0; ch < N_CHUNKS; ++ch) {
      se += ws_se[ch * SIMCLR_N + row];
      pd += ws_pos[ch * SIMCLR_N + row];
    }
    acc += __builtin_amdgcn_logf(se) - pd;   // log2(sumexp) - d_pos
  }
#pragma unroll
  for (int off = 16; off > 0; off >>= 1)
    acc += __shfl_xor(acc, off, 32);
  if ((threadIdx.x & 31) == 0) red[threadIdx.x >> 5] = acc;
  __syncthreads();
  if (threadIdx.x == 0) {
    float t = 0.0f;
#pragma unroll
    for (int i = 0; i < 8; ++i) t += red[i];
    *out = t * (0.69314718055994530942f / (float)SIMCLR_N);  // * ln2 / N
  }
}

// ---------------------------------------------------------------------------
extern "C" void kernel_launch(void* const* d_in, const int* in_sizes, int n_in,
                              void* d_out, int out_size, void* d_ws,
                              size_t ws_size, hipStream_t stream) {
  const float* zi = (const float*)d_in[0];
  const float* zj = (const float*)d_in[1];

  _Float16* zn = (_Float16*)d_ws;   // 8192*256*2 = 4 MB
  char* p = (char*)d_ws + (size_t)SIMCLR_N * SIMCLR_D * sizeof(_Float16);
  float* ws_se  = (float*)p;
  float* ws_pos = (float*)(p + sizeof(float) * N_CHUNKS * SIMCLR_N);
  float* out = (float*)d_out;

  simclr_normalize<<<SIMCLR_N / 8, 256, 0, stream>>>(zi, zj, zn);
  // 32 rowGroup-blocks * 4 chunks = 128 blocks of 8 waves.
  simclr_main<<<(SIMCLR_N / 32 / 8) * N_CHUNKS, 256, 0, stream>>>(
      zn, ws_se, ws_pos);
  simclr_finalize<<<1, 256, 0, stream>>>(ws_se, ws_pos, out);
}
